// KeypointWeighting_33285996544161
// MI455X (gfx1250) — matrix operand order repaired and checked by reading, not compiled, not run on hardware
//
#include <hip/hip_runtime.h>
#include <hip/hip_bf16.h>

typedef __attribute__((ext_vector_type(16))) __bf16 v16bf;
typedef __attribute__((ext_vector_type(8)))  __bf16 v8bf;
typedef __attribute__((ext_vector_type(4)))  __bf16 v4bf;
typedef __attribute__((ext_vector_type(8)))  float  v8f;

#define B_   8
#define N_   16384
#define C_   640
#define H_   256
#define O_   128
#define KT_  (C_/32)      // 20 k-steps of 32
#define THRESH_ 0.4f

// ---------------------------------------------------------------------------
// Pack fp32 weights into bf16 WMMA B-fragments.
// Fragment (nt, kt): 32 lanes x 16 bf16, lane L: column N = nt*16 + (L%16),
// K = kt*32 + (L<16 ? 0..15 : 16..31), K ascending within the 16 elements.
// frag index f = nt*KT_ + kt so the k-loop streams contiguously.
// ---------------------------------------------------------------------------
__global__ __launch_bounds__(256) void pack_weights(const float* __restrict__ w1,
                                                    const float* __restrict__ wo,
                                                    __bf16* __restrict__ pw1,
                                                    __bf16* __restrict__ pwo) {
    const int id   = blockIdx.x * blockDim.x + threadIdx.x;
    const int NF1  = (H_/16) * KT_;   // 320 fragments for w1
    const int NF2  = (O_/16) * KT_;   // 160 fragments for wo
    if (id < NF1 * 32) {
        const int f = id >> 5, lane = id & 31;
        const int nt = f / KT_, kt = f % KT_;
        const int n  = nt * 16 + (lane & 15);
        const int kb = kt * 32 + ((lane & 16) ? 16 : 0);
        __bf16* dst = pw1 + (size_t)f * 512 + (size_t)lane * 16;
        #pragma unroll
        for (int j = 0; j < 16; ++j)
            dst[j] = (__bf16)w1[(size_t)(kb + j) * H_ + n];
    } else if (id < (NF1 + NF2) * 32) {
        const int id2 = id - NF1 * 32;
        const int f = id2 >> 5, lane = id2 & 31;
        const int nt = f / KT_, kt = f % KT_;
        const int n  = nt * 16 + (lane & 15);
        const int kb = kt * 32 + ((lane & 16) ? 16 : 0);
        __bf16* dst = pwo + (size_t)f * 512 + (size_t)lane * 16;
        #pragma unroll
        for (int j = 0; j < 16; ++j)
            dst[j] = (__bf16)wo[(size_t)(kb + j) * O_ + n];
    }
}

__global__ __launch_bounds__(256) void zero_out(float4* __restrict__ p, int n4) {
    int i = blockIdx.x * blockDim.x + threadIdx.x;
    const float4 z = make_float4(0.f, 0.f, 0.f, 0.f);
    for (; i < n4; i += gridDim.x * blockDim.x) p[i] = z;
}

// ---------------------------------------------------------------------------
// Shared helper: stage a 64x640 fp32 tile as bf16 into padded LDS.
// ---------------------------------------------------------------------------
__device__ __forceinline__ void stage_tile(const float* __restrict__ Xg,
                                           __bf16 (*Xs)[648], int t) {
    #pragma unroll
    for (int i = 0; i < 40; ++i) {
        const int idx4 = i * 256 + t;           // 64 rows * 160 float4/row
        const int row  = idx4 / 160;
        const int col  = (idx4 % 160) * 4;
        const float4 v = *(const float4*)(Xg + (size_t)row * C_ + col);
        v4bf bv;
        bv[0] = (__bf16)v.x; bv[1] = (__bf16)v.y;
        bv[2] = (__bf16)v.z; bv[3] = (__bf16)v.w;
        *(v4bf*)(&Xs[row][col]) = bv;
    }
}

__device__ __forceinline__ v16bf load_afrag(const __bf16 (*Xs)[648], int m,
                                            int c0, int ksel) {
    v8bf lo = *(const v8bf*)(&Xs[m][c0 + ksel]);
    v8bf hi = *(const v8bf*)(&Xs[m][c0 + 8 + ksel]);
    return __builtin_shufflevector(lo, hi, 0,1,2,3,4,5,6,7,8,9,10,11,12,13,14,15);
}

// ---------------------------------------------------------------------------
// Kernel 1: h = relu(X@W1 + b1); logit = h@w2 + b2; score = sigmoid(logit)
// 64 points/block, 8 waves: wave w -> point subtile (w&3), H-half (w>>2).
// ---------------------------------------------------------------------------
__global__ __launch_bounds__(256) void mlp_scores(const float* __restrict__ feat,
                                                  const __bf16* __restrict__ pw1,
                                                  const float* __restrict__ b1,
                                                  const float* __restrict__ w2,
                                                  const float* __restrict__ b2,
                                                  float* __restrict__ scores) {
    __shared__ __bf16 Xs[64][648];
    __shared__ float  lgt[64];
    const int t    = threadIdx.x;
    const int lane = t & 31, w = t >> 5;
    const int b    = blockIdx.x >> 8;          // 256 tiles per batch
    const int n0   = (blockIdx.x & 255) << 6;  // *64

    stage_tile(feat + ((size_t)b * N_ + n0) * C_, Xs, t);
    if (t < 64) lgt[t] = 0.f;
    __syncthreads();

    const int p    = w & 3;
    const int hh   = w >> 2;
    const int m    = p * 16 + (lane & 15);
    const int ksel = (lane & 16) ? 16 : 0;

    v8f acc[8];
    const v8f z8 = {0.f,0.f,0.f,0.f,0.f,0.f,0.f,0.f};
    #pragma unroll
    for (int j = 0; j < 8; ++j) acc[j] = z8;

    for (int kt = 0; kt < KT_; ++kt) {
        const v16bf a = load_afrag(Xs, m, kt * 32, ksel);
        #pragma unroll
        for (int j = 0; j < 8; ++j) {
            const int ht = hh * 8 + j;
            const v16bf bb = *(const v16bf*)(pw1 + ((size_t)(ht * KT_ + kt)) * 512
                                                  + (size_t)lane * 16);
            acc[j] = __builtin_amdgcn_wmma_f32_16x16x32_bf16(
                false, a, false, bb, (short)0, acc[j], false, false);
        }
    }

    // epilogue: bias + relu + dot with w2, reduce into per-point logits
    float partial[8] = {0.f,0.f,0.f,0.f,0.f,0.f,0.f,0.f};
    const int nl = lane & 15;
    #pragma unroll
    for (int j = 0; j < 8; ++j) {
        const int nh = (hh * 8 + j) * 16 + nl;
        const float w2v = w2[nh], b1v = b1[nh];
        #pragma unroll
        for (int r = 0; r < 8; ++r) {
            float h = acc[j][r] + b1v;
            h = h > 0.f ? h : 0.f;
            partial[r] += h * w2v;
        }
    }
    const int roff = (lane & 16) ? 8 : 0;
    #pragma unroll
    for (int r = 0; r < 8; ++r)
        atomicAdd(&lgt[p * 16 + roff + r], partial[r]);
    __syncthreads();

    if (t < 64) {
        const float lg = lgt[t] + b2[0];
        scores[(size_t)b * N_ + n0 + t] = 1.f / (1.f + __expf(-lg));
    }
}

// ---------------------------------------------------------------------------
// Kernel 2: per-batch exclusive scan of (score > THRESH) -> destination slot
// ---------------------------------------------------------------------------
__global__ __launch_bounds__(1024) void scan_pos(const float* __restrict__ scores,
                                                 int* __restrict__ pos) {
    __shared__ int s[1024];
    const int b = blockIdx.x, t = threadIdx.x;
    const float* sc = scores + (size_t)b * N_;
    int* pp = pos + (size_t)b * N_;
    const int base = t * 16;
    unsigned flags = 0; int sum = 0;
    #pragma unroll
    for (int i = 0; i < 16; ++i) {
        const bool sel = sc[base + i] > THRESH_;
        flags |= (sel ? 1u : 0u) << i;
        sum += sel ? 1 : 0;
    }
    s[t] = sum;
    __syncthreads();
    for (int off = 1; off < 1024; off <<= 1) {
        const int v = (t >= off) ? s[t - off] : 0;
        __syncthreads();
        s[t] += v;
        __syncthreads();
    }
    int excl = s[t] - sum;
    #pragma unroll
    for (int i = 0; i < 16; ++i) {
        const bool sel = (flags >> i) & 1u;
        pp[base + i] = sel ? excl : -1;
        if (sel) excl++;
    }
}

// ---------------------------------------------------------------------------
// Kernel 3: feats = score * (X@Wo) + bo, scattered to compacted slots;
// coords scattered too. Wave w -> subtile (w&3), O-tiles (w>>2)*4..+3.
// ---------------------------------------------------------------------------
__global__ __launch_bounds__(256) void proj_scatter(const float* __restrict__ feat,
                                                    const float* __restrict__ coords,
                                                    const __bf16* __restrict__ pwo,
                                                    const float* __restrict__ bo,
                                                    const float* __restrict__ scores,
                                                    const int* __restrict__ pos,
                                                    float* __restrict__ kp_out,
                                                    float* __restrict__ co_out) {
    __shared__ __bf16 Xs[64][648];
    __shared__ float s_l[64];
    __shared__ int   p_l[64];
    const int t    = threadIdx.x;
    const int lane = t & 31, w = t >> 5;
    const int b    = blockIdx.x >> 8;
    const int n0   = (blockIdx.x & 255) << 6;

    stage_tile(feat + ((size_t)b * N_ + n0) * C_, Xs, t);
    if (t < 64) {
        s_l[t] = scores[(size_t)b * N_ + n0 + t];
        p_l[t] = pos[(size_t)b * N_ + n0 + t];
    }
    __syncthreads();

    const int p    = w & 3;
    const int og   = w >> 2;
    const int m    = p * 16 + (lane & 15);
    const int ksel = (lane & 16) ? 16 : 0;

    v8f acc[4];
    const v8f z8 = {0.f,0.f,0.f,0.f,0.f,0.f,0.f,0.f};
    #pragma unroll
    for (int j = 0; j < 4; ++j) acc[j] = z8;

    for (int kt = 0; kt < KT_; ++kt) {
        const v16bf a = load_afrag(Xs, m, kt * 32, ksel);
        #pragma unroll
        for (int j = 0; j < 4; ++j) {
            const int ot = og * 4 + j;
            const v16bf bb = *(const v16bf*)(pwo + ((size_t)(ot * KT_ + kt)) * 512
                                                  + (size_t)lane * 16);
            acc[j] = __builtin_amdgcn_wmma_f32_16x16x32_bf16(
                false, a, false, bb, (short)0, acc[j], false, false);
        }
    }

    // epilogue: scale by score, add bias, scatter rows to compacted slots
    const int ol   = lane & 15;
    const int roff = (lane & 16) ? 8 : 0;
    #pragma unroll
    for (int j = 0; j < 4; ++j) {
        const int o = (og * 4 + j) * 16 + ol;
        const float bov = bo[o];
        #pragma unroll
        for (int r = 0; r < 8; ++r) {
            const int mr = p * 16 + roff + r;
            const int dp = p_l[mr];
            if (dp >= 0)
                kp_out[((size_t)b * N_ + dp) * O_ + o] = acc[j][r] * s_l[mr] + bov;
        }
    }
    if (t < 64) {
        const int dp = p_l[t];
        if (dp >= 0) {
            const float* cs = coords + ((size_t)b * N_ + n0 + t) * 3;
            float* cd = co_out + ((size_t)b * N_ + dp) * 3;
            cd[0] = cs[0]; cd[1] = cs[1]; cd[2] = cs[2];
        }
    }
}

// point_mask: True where any coordinate of the (compacted, zero-padded) row == 0
__global__ __launch_bounds__(256) void mask_kernel(const float* __restrict__ co,
                                                   float* __restrict__ mk) {
    const int i = blockIdx.x * blockDim.x + threadIdx.x;
    if (i < B_ * N_) {
        const float x = co[(size_t)i * 3 + 0];
        const float y = co[(size_t)i * 3 + 1];
        const float z = co[(size_t)i * 3 + 2];
        mk[i] = (x == 0.f || y == 0.f || z == 0.f) ? 1.f : 0.f;
    }
}

// ---------------------------------------------------------------------------
extern "C" void kernel_launch(void* const* d_in, const int* in_sizes, int n_in,
                              void* d_out, int out_size, void* d_ws, size_t ws_size,
                              hipStream_t stream) {
    const float* feat   = (const float*)d_in[0];
    const float* coords = (const float*)d_in[1];
    const float* w1     = (const float*)d_in[2];
    const float* b1     = (const float*)d_in[3];
    const float* w2     = (const float*)d_in[4];
    const float* b2     = (const float*)d_in[5];
    const float* wo     = (const float*)d_in[6];
    const float* bo     = (const float*)d_in[7];

    float* out = (float*)d_out;
    float* kp  = out;                                    // [8,16384,128]
    float* co  = kp + (size_t)B_ * N_ * O_;              // [8,16384,3]
    float* mk  = co + (size_t)B_ * N_ * 3;               // [8,16384]

    char* ws = (char*)d_ws;
    __bf16* pw1   = (__bf16*)ws;                          // 327,680 B
    __bf16* pwo   = (__bf16*)(ws + 327680);               // 163,840 B
    float*  scor  = (float*)(ws + 327680 + 163840);       // 524,288 B
    int*    pos   = (int*)(ws + 327680 + 163840 + 524288);// 524,288 B

    pack_weights<<<60, 256, 0, stream>>>(w1, wo, pw1, pwo);
    const int n4 = (int)(((size_t)B_ * N_ * (O_ + 3)) / 4);
    zero_out<<<2048, 256, 0, stream>>>((float4*)kp, n4);
    mlp_scores<<<B_ * (N_ / 64), 256, 0, stream>>>(feat, pw1, b1, w2, b2, scor);
    scan_pos<<<B_, 1024, 0, stream>>>(scor, pos);
    proj_scatter<<<B_ * (N_ / 64), 256, 0, stream>>>(feat, coords, pwo, bo, scor,
                                                     pos, kp, co);
    mask_kernel<<<(B_ * N_ + 255) / 256, 256, 0, stream>>>(co, mk);
}
